// LAHRv2_40192303956203
// MI455X (gfx1250) — compile-verified, matrix operand order checked by reading, not run on hardware
//
#include <hip/hip_runtime.h>
#include <hip/hip_bf16.h>
#include <math.h>

// ---------------- model constants ----------------
static constexpr int Bc   = 2;
static constexpr int Tc   = 1024;
static constexpr int Dc   = 1024;
static constexpr int Hc   = 16;
static constexpr int Lc   = 12;
static constexpr int Vc   = 50257;
static constexpr int DFFc = 2730;
static constexpr int CAPc = 128;
static constexpr int NSLOTSc = 128;
static constexpr int TOPKc   = 8;
static constexpr int NTHc    = 4;
static constexpr int Mfull   = Bc * Tc;    // 2048 token rows
static constexpr int Mcap    = Bc * CAPc;  // 256 routed rows

// ---------------- vector types ----------------
typedef __attribute__((ext_vector_type(16))) __bf16 bf16x16;
typedef __attribute__((ext_vector_type(16))) short  shortx16;
typedef __attribute__((ext_vector_type(8)))  float  floatx8;
typedef __attribute__((ext_vector_type(4)))  unsigned short ushortx4;
typedef __attribute__((ext_vector_type(4)))  unsigned uintx4;
typedef __attribute__((ext_vector_type(4)))  int      intx4;
typedef __attribute__((ext_vector_type(8)))  int      intx8;

__device__ __forceinline__ unsigned short f2bf(float f) {
  unsigned u = __builtin_bit_cast(unsigned, f);
  unsigned r = u + 0x7FFFu + ((u >> 16) & 1u);   // round-to-nearest-even
  return (unsigned short)(r >> 16);
}

__device__ __forceinline__ float act_apply(float v, int act) {
  if (act == 1) return v / (1.f + __expf(-v));                    // silu
  if (act == 2) return 0.5f * v * (1.f + erff(v * 0.70710678f));  // exact gelu
  if (act == 3) return 1.f / (1.f + __expf(-v));                  // sigmoid
  return v;
}

// ---------------- Tensor Data Mover: 2D tile global->LDS ----------------
// Builds a D# per CDNA5 ISA (group0: count/lds/global/type,
// group1: data_size=4B, LDS pad every 32 DWORDs by 4 DWORDs, tensor dims for
// OOB-zeroing, dim0 stride). 2D tile: groups 2/3 unused (zeros).
__device__ __forceinline__ void tdm_load_2d(unsigned ldsOff, const void* gptr,
                                            int tileCols, int tileRows,
                                            int remCols, int remRows, int strideElems) {
  unsigned long long ga = (unsigned long long)gptr;
  uintx4 g0;
  g0[0] = 1u;                                               // count=1 (valid user D#)
  g0[1] = ldsOff;                                           // lds_addr (bytes)
  g0[2] = (unsigned)(ga & 0xFFFFFFFFull);                   // global_addr[31:0]
  g0[3] = (unsigned)((ga >> 32) & 0x1FFFFFFull) | (2u << 30); // global_addr[56:32] | type=2
  const unsigned d0 = (2u << 16)    // data_size = 4 bytes
                    | (1u << 20)    // pad_enable
                    | (4u << 22)    // pad_interval: every 32 DWORDs
                    | (3u << 25);   // pad_amount: 4 DWORDs
  unsigned td0 = (unsigned)remCols;
  unsigned td1 = (unsigned)remRows;
  intx8 g1;
  g1[0] = (int)d0;
  g1[1] = (int)((td0 & 0xFFFFu) << 16);                     // tensor_dim0[15:0]
  g1[2] = (int)((td0 >> 16) | ((td1 & 0xFFFFu) << 16));     // dim0 hi | dim1 lo
  g1[3] = (int)((td1 >> 16) | ((unsigned)tileCols << 16));  // dim1 hi | tile_dim0
  g1[4] = (int)((unsigned)tileRows & 0xFFFFu);              // tile_dim1 (tile_dim2=0)
  g1[5] = (int)(unsigned)strideElems;                       // tensor_dim0_stride lo
  g1[6] = 0;
  g1[7] = 0;
  intx4 g2 = {0, 0, 0, 0};
  intx4 g3 = {0, 0, 0, 0};
#if defined(__clang_major__) && __clang_major__ >= 23
  intx8 g4 = {0, 0, 0, 0, 0, 0, 0, 0};
  __builtin_amdgcn_tensor_load_to_lds(g0, g1, g2, g3, g4, 0);
#else
  __builtin_amdgcn_tensor_load_to_lds(g0, g1, g2, g3, 0);
#endif
}

// ---------------- TDM / cluster probe kernel (kept from round 2) ----------------
__global__ void k_tdm_touch() {
  __shared__ unsigned dummy[32];
  if (threadIdx.x == 0) {
    uintx4 g0 = {0u, 0u, 0u, 0u};          // count=0 => NULL tensor descriptor
    intx8  g1 = {0, 0, 0, 0, 0, 0, 0, 0};
    intx4  g2 = {0, 0, 0, 0};
    intx4  g3 = {0, 0, 0, 0};
#if defined(__clang_major__) && __clang_major__ >= 23
    intx8  g4 = {0, 0, 0, 0, 0, 0, 0, 0};
    __builtin_amdgcn_tensor_load_to_lds(g0, g1, g2, g3, g4, 0);
#else
    __builtin_amdgcn_tensor_load_to_lds(g0, g1, g2, g3, 0);
#endif
    __builtin_amdgcn_s_wait_tensorcnt(0);
    dummy[0] = (unsigned)__builtin_amdgcn_cluster_id_x();
  }
  __builtin_amdgcn_s_cluster_barrier();
  dummy[threadIdx.x & 31] += threadIdx.x;
}

// ---------------- GEMM tiling constants ----------------
// 128x64 tile / WG; 8 waves, each wave owns a 32x32 slab:
//   2 A-frags x 2 B-frags -> 4x v_wmma_f32_16x16x32_bf16 per K-step (BK=32).
static constexpr int GBM = 128, GBN = 64, GBK = 32, GLDS = 40; // +8 ushort pad
static constexpr int TST = 36;   // TDM-staged f32 row stride (32 data + 4 pad DWORDs)

// ---------------- generic GEMM: C = act(alpha * A[M,K] @ W[N,K]^T + bias) ----------
__global__ __launch_bounds__(256) void k_gemm_bf16(
    const float* __restrict__ A, const float* __restrict__ W,
    const float* __restrict__ bias, float* __restrict__ C,
    int M, int N, int K, float alpha, int act) {
  __shared__ unsigned short As[GBM][GLDS];
  __shared__ unsigned short Bs[GBN][GLDS];

  const int tid  = threadIdx.x;
  const int lane = tid & 31;
  const int wave = tid >> 5;        // 0..7
  const int wm   = wave >> 1;       // 0..3 : 32-row slab
  const int wn   = wave & 1;        // 0..1 : 32-col slab
  const int hlf  = lane >> 4;       // half-wave
  const int r    = lane & 15;

  const int blockM = blockIdx.y * GBM;
  const int blockN = blockIdx.x * GBN;
  const bool v4ok  = ((K & 3) == 0);

  floatx8 acc[2][2] = { { {0,0,0,0,0,0,0,0}, {0,0,0,0,0,0,0,0} },
                        { {0,0,0,0,0,0,0,0}, {0,0,0,0,0,0,0,0} } };

  for (int k0 = 0; k0 < K; k0 += GBK) {
    if (k0 + GBK < K) {
      int pr = blockM + (tid >> 1);
      if (pr < M) __builtin_prefetch(&A[(size_t)pr * K + k0 + GBK + ((tid & 1) << 4)], 0, 1);
      int pb = blockN + (tid >> 2);
      if (pb < N) __builtin_prefetch(&W[(size_t)pb * K + k0 + GBK + ((tid & 3) << 3)], 0, 1);
    }
    // ---- stage A tile (128x32) f32 -> bf16 ----
    for (int i = tid; i < (GBM * GBK) / 4; i += 256) {
      int row = i >> 3, c4 = (i & 7) << 2;
      int gr = blockM + row;
      float4 v = make_float4(0.f, 0.f, 0.f, 0.f);
      if (gr < M) {
        if (v4ok && (k0 + GBK <= K)) {
          v = *(const float4*)(A + (size_t)gr * K + k0 + c4);
        } else {
          float t0 = 0.f, t1 = 0.f, t2 = 0.f, t3 = 0.f;
          int gc = k0 + c4;
          if (gc + 0 < K) t0 = A[(size_t)gr * K + gc + 0];
          if (gc + 1 < K) t1 = A[(size_t)gr * K + gc + 1];
          if (gc + 2 < K) t2 = A[(size_t)gr * K + gc + 2];
          if (gc + 3 < K) t3 = A[(size_t)gr * K + gc + 3];
          v = make_float4(t0, t1, t2, t3);
        }
      }
      ushortx4 p;
      p[0] = f2bf(v.x); p[1] = f2bf(v.y); p[2] = f2bf(v.z); p[3] = f2bf(v.w);
      *(ushortx4*)&As[row][c4] = p;
    }
    // ---- stage W tile (64x32) ----
    for (int i = tid; i < (GBN * GBK) / 4; i += 256) {
      int row = i >> 3, c4 = (i & 7) << 2;
      int gr = blockN + row;
      float4 v = make_float4(0.f, 0.f, 0.f, 0.f);
      if (gr < N) {
        if (v4ok && (k0 + GBK <= K)) {
          v = *(const float4*)(W + (size_t)gr * K + k0 + c4);
        } else {
          float t0 = 0.f, t1 = 0.f, t2 = 0.f, t3 = 0.f;
          int gc = k0 + c4;
          if (gc + 0 < K) t0 = W[(size_t)gr * K + gc + 0];
          if (gc + 1 < K) t1 = W[(size_t)gr * K + gc + 1];
          if (gc + 2 < K) t2 = W[(size_t)gr * K + gc + 2];
          if (gc + 3 < K) t3 = W[(size_t)gr * K + gc + 3];
          v = make_float4(t0, t1, t2, t3);
        }
      }
      ushortx4 p;
      p[0] = f2bf(v.x); p[1] = f2bf(v.y); p[2] = f2bf(v.z); p[3] = f2bf(v.w);
      *(ushortx4*)&Bs[row][c4] = p;
    }
    __syncthreads();

    shortx16 araw[2], braw[2];
    #pragma unroll
    for (int ai = 0; ai < 2; ++ai) {
      const int aRow = wm * 32 + ai * 16 + r;
      #pragma unroll
      for (int g = 0; g < 2; ++g)
        #pragma unroll
        for (int e = 0; e < 8; ++e)
          araw[ai][g * 8 + e] = (short)As[aRow][g * 16 + hlf * 8 + e];
    }
    #pragma unroll
    for (int bi = 0; bi < 2; ++bi) {
      const int bRow = wn * 32 + bi * 16 + r;
      #pragma unroll
      for (int g = 0; g < 2; ++g)
        #pragma unroll
        for (int e = 0; e < 8; ++e)
          braw[bi][g * 8 + e] = (short)Bs[bRow][g * 16 + hlf * 8 + e];
    }
    bf16x16 af[2] = { __builtin_bit_cast(bf16x16, araw[0]),
                      __builtin_bit_cast(bf16x16, araw[1]) };
    bf16x16 bf[2] = { __builtin_bit_cast(bf16x16, braw[0]),
                      __builtin_bit_cast(bf16x16, braw[1]) };

    #pragma unroll
    for (int ai = 0; ai < 2; ++ai)
      #pragma unroll
      for (int bi = 0; bi < 2; ++bi)
        acc[ai][bi] = __builtin_amdgcn_wmma_f32_16x16x32_bf16(
            false, af[ai], false, bf[bi], (short)0, acc[ai][bi], false, false);
    __syncthreads();
  }

  #pragma unroll
  for (int ai = 0; ai < 2; ++ai) {
    const int rowBase = blockM + wm * 32 + ai * 16 + hlf * 8;
    #pragma unroll
    for (int bi = 0; bi < 2; ++bi) {
      int col = blockN + wn * 32 + bi * 16 + r;
      if (col < N) {
        float bv = bias ? bias[col] : 0.f;
        #pragma unroll
        for (int rr = 0; rr < 8; ++rr) {
          int row = rowBase + rr;
          if (row < M)
            C[(size_t)row * N + col] = act_apply(acc[ai][bi][rr] * alpha + bv, act);
        }
      }
    }
  }
}

// ---------------- TDM-staged GEMM (used for the LM-head GEMM) ----------------
// Double-buffered f32 staging tiles filled by the Tensor Data Mover; wave 0
// issues tensor_load_to_lds for the next K-slab and gates buffers with
// s_wait_tensorcnt + workgroup barrier. Fragments are converted f32->bf16
// on the fly while feeding 4 WMMAs per K-step per wave.
__global__ __launch_bounds__(256) void k_gemm_bf16_tdm(
    const float* __restrict__ A, const float* __restrict__ W,
    float* __restrict__ C, int M, int N, int K, float alpha) {
  __shared__ float Ast[2][GBM * TST];   // 2 x 128 x 36 f32 = 36 KB
  __shared__ float Bst[2][GBN * TST];   // 2 x  64 x 36 f32 = 18 KB

  const int tid  = threadIdx.x;
  const int lane = tid & 31;
  const int wave = tid >> 5;
  const int wm   = wave >> 1;
  const int wn   = wave & 1;
  const int hlf  = lane >> 4;
  const int r    = lane & 15;

  const int blockM = blockIdx.y * GBM;
  const int blockN = blockIdx.x * GBN;

  floatx8 acc[2][2] = { { {0,0,0,0,0,0,0,0}, {0,0,0,0,0,0,0,0} },
                        { {0,0,0,0,0,0,0,0}, {0,0,0,0,0,0,0,0} } };

  if (wave == 0) {
    tdm_load_2d((unsigned)(unsigned long long)(const void*)&Ast[0][0],
                A + (size_t)blockM * K, GBK, GBM, K, M - blockM, K);
    tdm_load_2d((unsigned)(unsigned long long)(const void*)&Bst[0][0],
                W + (size_t)blockN * K, GBK, GBN, K, N - blockN, K);
  }

  int cur = 0;
  for (int k0 = 0; k0 < K; k0 += GBK, cur ^= 1) {
    if (wave == 0) __builtin_amdgcn_s_wait_tensorcnt(0);
    __syncthreads();                       // buffer `cur` ready; prev reads done
    if (wave == 0 && (k0 + GBK < K)) {     // kick DMA for the next K-slab
      const int nk = k0 + GBK;
      tdm_load_2d((unsigned)(unsigned long long)(const void*)&Ast[cur ^ 1][0],
                  A + (size_t)blockM * K + nk, GBK, GBM, K - nk, M - blockM, K);
      tdm_load_2d((unsigned)(unsigned long long)(const void*)&Bst[cur ^ 1][0],
                  W + (size_t)blockN * K + nk, GBK, GBN, K - nk, N - blockN, K);
    }

    const float* Ab = &Ast[cur][0];
    const float* Bb = &Bst[cur][0];
    shortx16 araw[2], braw[2];
    #pragma unroll
    for (int ai = 0; ai < 2; ++ai) {
      const int aRow = wm * 32 + ai * 16 + r;
      #pragma unroll
      for (int g = 0; g < 2; ++g)
        #pragma unroll
        for (int e = 0; e < 8; ++e)
          araw[ai][g * 8 + e] = (short)f2bf(Ab[aRow * TST + g * 16 + hlf * 8 + e]);
    }
    #pragma unroll
    for (int bi = 0; bi < 2; ++bi) {
      const int bRow = wn * 32 + bi * 16 + r;
      #pragma unroll
      for (int g = 0; g < 2; ++g)
        #pragma unroll
        for (int e = 0; e < 8; ++e)
          braw[bi][g * 8 + e] = (short)f2bf(Bb[bRow * TST + g * 16 + hlf * 8 + e]);
    }
    bf16x16 af[2] = { __builtin_bit_cast(bf16x16, araw[0]),
                      __builtin_bit_cast(bf16x16, araw[1]) };
    bf16x16 bf[2] = { __builtin_bit_cast(bf16x16, braw[0]),
                      __builtin_bit_cast(bf16x16, braw[1]) };

    #pragma unroll
    for (int ai = 0; ai < 2; ++ai)
      #pragma unroll
      for (int bi = 0; bi < 2; ++bi)
        acc[ai][bi] = __builtin_amdgcn_wmma_f32_16x16x32_bf16(
            false, af[ai], false, bf[bi], (short)0, acc[ai][bi], false, false);
  }

  #pragma unroll
  for (int ai = 0; ai < 2; ++ai) {
    const int rowBase = blockM + wm * 32 + ai * 16 + hlf * 8;
    #pragma unroll
    for (int bi = 0; bi < 2; ++bi) {
      int col = blockN + wn * 32 + bi * 16 + r;
      if (col < N) {
        #pragma unroll
        for (int rr = 0; rr < 8; ++rr) {
          int row = rowBase + rr;
          if (row < M)
            C[(size_t)row * N + col] = acc[ai][bi][rr] * alpha;
        }
      }
    }
  }
}

// ---------------- embedding: x = embed[ids] + pos ----------------
__global__ void k_embed(const int* __restrict__ ids, const float* __restrict__ emb,
                        const float* __restrict__ pos, float* __restrict__ x, int T, int D) {
  int row = blockIdx.x;             // b*T + t
  int t = row % T;
  int id = ids[row];
  for (int d = threadIdx.x; d < D; d += blockDim.x)
    x[(size_t)row * D + d] = emb[(size_t)id * D + d] + pos[(size_t)t * D + d];
}

// ---------------- RMSNorm ----------------
__global__ __launch_bounds__(256) void k_rmsnorm(const float* __restrict__ x,
                                                 const float* __restrict__ w,
                                                 float* __restrict__ y, int D) {
  __shared__ float red[256];
  int row = blockIdx.x;
  const float* xr = x + (size_t)row * D;
  float s = 0.f;
  for (int d = threadIdx.x; d < D; d += 256) { float v = xr[d]; s += v * v; }
  red[threadIdx.x] = s; __syncthreads();
  for (int o = 128; o > 0; o >>= 1) {
    if (threadIdx.x < o) red[threadIdx.x] += red[threadIdx.x + o];
    __syncthreads();
  }
  float rstd = rsqrtf(red[0] / (float)D + 1e-6f);
  for (int d = threadIdx.x; d < D; d += 256)
    y[(size_t)row * D + d] = xr[d] * rstd * w[d];
}

// ---------------- causal attention, one WG per (b,h,q) ----------------
__global__ __launch_bounds__(128) void k_attn(const float* __restrict__ qkv,
                                              float* __restrict__ ctx,
                                              int T, int D, int H) {
  __shared__ float qv[64];
  __shared__ float sc[1024];
  __shared__ float red[128];
  const int qpos = blockIdx.x;
  const int b = blockIdx.y / H, h = blockIdx.y % H;
  const int hd = D / H;
  const int tid = threadIdx.x;
  const float* base = qkv + (size_t)b * T * 3 * D;
  const float* qrow = base + (size_t)qpos * 3 * D + h * hd;
  if (tid < hd) qv[tid] = qrow[tid];
  __syncthreads();
  const float scale = rsqrtf((float)hd);
  float lmax = -INFINITY;
  for (int k = tid; k <= qpos; k += 128) {
    const float* krow = base + (size_t)k * 3 * D + D + h * hd;
    float s = 0.f;
    for (int j = 0; j < 64; ++j) s += qv[j] * krow[j];
    s *= scale;
    sc[k] = s;
    lmax = fmaxf(lmax, s);
  }
  red[tid] = lmax; __syncthreads();
  for (int o = 64; o > 0; o >>= 1) {
    if (tid < o) red[tid] = fmaxf(red[tid], red[tid + o]);
    __syncthreads();
  }
  float m = red[0]; __syncthreads();
  float lsum = 0.f;
  for (int k = tid; k <= qpos; k += 128) {
    float e = __expf(sc[k] - m);
    sc[k] = e; lsum += e;
  }
  red[tid] = lsum; __syncthreads();
  for (int o = 64; o > 0; o >>= 1) {
    if (tid < o) red[tid] += red[tid + o];
    __syncthreads();
  }
  float inv = 1.f / red[0];
  if (tid < hd) {
    float o = 0.f;
    for (int k = 0; k <= qpos; ++k) {
      const float* vrow = base + (size_t)k * 3 * D + 2 * D + h * hd;
      o += sc[k] * vrow[tid];
    }
    ctx[((size_t)(b * T + qpos)) * D + h * hd + tid] = o * inv;
  }
}

// ---------------- router score = dot(x_row, rw) ----------------
__global__ __launch_bounds__(256) void k_router(const float* __restrict__ x,
                                                const float* __restrict__ rw,
                                                float* __restrict__ scores, int D) {
  __shared__ float red[256];
  int row = blockIdx.x;
  float s = 0.f;
  for (int d = threadIdx.x; d < D; d += 256) s += x[(size_t)row * D + d] * rw[d];
  red[threadIdx.x] = s; __syncthreads();
  for (int o = 128; o > 0; o >>= 1) {
    if (threadIdx.x < o) red[threadIdx.x] += red[threadIdx.x + o];
    __syncthreads();
  }
  if (threadIdx.x == 0) scores[row] = red[0];
}

// ---------------- exact top-CAP selection (descending, stable ties) ----------------
__global__ __launch_bounds__(256) void k_topk_tokens(const float* __restrict__ scores,
                                                     int* __restrict__ idx, int T, int CAP) {
  __shared__ float s[1024];
  int b = blockIdx.x;
  for (int i = threadIdx.x; i < T; i += 256) s[i] = scores[b * T + i];
  __syncthreads();
  for (int i = threadIdx.x; i < T; i += 256) {
    float si = s[i];
    int rank = 0;
    for (int j = 0; j < T; ++j) {
      float sj = s[j];
      rank += (sj > si) || (sj == si && j < i);
    }
    if (rank < CAP) idx[b * CAP + rank] = i;
  }
}

__global__ void k_gather_rows(const float* __restrict__ x, const int* __restrict__ idx,
                              float* __restrict__ xs, int T, int CAP, int D) {
  int row = blockIdx.x;                 // b*CAP + c
  int b = row / CAP;
  int src = b * T + idx[row];
  for (int d = threadIdx.x; d < D; d += blockDim.x)
    xs[(size_t)row * D + d] = x[(size_t)src * D + d];
}

__global__ void k_scatter_rows(float* __restrict__ x, const int* __restrict__ idx,
                               const float* __restrict__ xs, int T, int CAP, int D) {
  int row = blockIdx.x;
  int b = row / CAP;
  int dst = b * T + idx[row];
  for (int d = threadIdx.x; d < D; d += blockDim.x)
    x[(size_t)dst * D + d] = xs[(size_t)row * D + d];
}

// ---------------- kNN memory: per-token top-8 + softmax weights ----------------
__global__ void k_memtopk(const float* __restrict__ sim, float* __restrict__ tkw,
                          int* __restrict__ tki, int Mrows, int nslots, int topk) {
  int row = blockIdx.x * blockDim.x + threadIdx.x;
  if (row >= Mrows) return;
  const float* s = sim + (size_t)row * nslots;
  float vals[TOPKc]; int ids[TOPKc];
  unsigned long long tl = 0, th = 0;
  for (int k = 0; k < topk; ++k) {
    float best = -INFINITY; int bi = 0;
    for (int mm = 0; mm < nslots; ++mm) {
      bool taken = (mm < 64) ? ((tl >> mm) & 1ull) : ((th >> (mm - 64)) & 1ull);
      if (!taken && s[mm] > best) { best = s[mm]; bi = mm; }
    }
    vals[k] = best; ids[k] = bi;
    if (bi < 64) tl |= 1ull << bi; else th |= 1ull << (bi - 64);
  }
  float m0 = vals[0], sum = 0.f;
  for (int k = 0; k < topk; ++k) { vals[k] = __expf(vals[k] - m0); sum += vals[k]; }
  float inv = 1.f / sum;
  for (int k = 0; k < topk; ++k) {
    tkw[(size_t)row * topk + k] = vals[k] * inv;
    tki[(size_t)row * topk + k] = ids[k];
  }
}

__global__ void k_retrieve(const float* __restrict__ tkw, const int* __restrict__ tki,
                           const float* __restrict__ memv, float* __restrict__ rv,
                           int D, int topk) {
  int row = blockIdx.x;
  for (int d = threadIdx.x; d < D; d += blockDim.x) {
    float a = 0.f;
    for (int k = 0; k < topk; ++k)
      a += tkw[(size_t)row * topk + k] * memv[(size_t)tki[(size_t)row * topk + k] * D + d];
    rv[(size_t)row * D + d] = a;
  }
}

__global__ void k_concat(const float* __restrict__ a, const float* __restrict__ b,
                         float* __restrict__ out, int D) {
  int row = blockIdx.x;
  for (int d = threadIdx.x; d < D; d += blockDim.x) {
    out[(size_t)row * 2 * D + d]     = a[(size_t)row * D + d];
    out[(size_t)row * 2 * D + D + d] = b[(size_t)row * D + d];
  }
}

__global__ void k_gatefuse(float* __restrict__ x, const float* __restrict__ gate,
                           const float* __restrict__ retr, int D) {
  int row = blockIdx.x;
  float g = gate[row];
  for (int d = threadIdx.x; d < D; d += blockDim.x)
    x[(size_t)row * D + d] += g * retr[(size_t)row * D + d];
}

__global__ void k_add(float* __restrict__ y, const float* __restrict__ x, size_t n) {
  size_t i = (size_t)blockIdx.x * blockDim.x + threadIdx.x;
  size_t stride = (size_t)gridDim.x * blockDim.x;
  for (; i < n; i += stride) y[i] += x[i];
}

__global__ void k_mul(float* __restrict__ y, const float* __restrict__ x, size_t n) {
  size_t i = (size_t)blockIdx.x * blockDim.x + threadIdx.x;
  size_t stride = (size_t)gridDim.x * blockDim.x;
  for (; i < n; i += stride) y[i] *= x[i];
}

// =====================================================================
extern "C" void kernel_launch(void* const* d_in, const int* in_sizes, int n_in,
                              void* d_out, int out_size, void* d_ws, size_t ws_size,
                              hipStream_t stream) {
  const int*   ids      = (const int*)  d_in[0];
  const float* embed    = (const float*)d_in[1];
  const float* pos      = (const float*)d_in[2];
  const float* n1       = (const float*)d_in[3];
  const float* qkv_w    = (const float*)d_in[4];
  const float* out_w    = (const float*)d_in[5];
  const float* n2       = (const float*)d_in[6];
  const float* w1       = (const float*)d_in[7];
  const float* w2       = (const float*)d_in[8];
  const float* w3       = (const float*)d_in[9];
  const float* router   = (const float*)d_in[10];
  const float* mem_keys = (const float*)d_in[11];
  const float* mem_vals = (const float*)d_in[12];
  const float* mem_q    = (const float*)d_in[13];
  const float* mem_out  = (const float*)d_in[14];
  const float* gate_w1  = (const float*)d_in[15];
  const float* gate_b1  = (const float*)d_in[16];
  const float* gate_w2  = (const float*)d_in[17];
  const float* gate_b2  = (const float*)d_in[18];
  const float* lat_norm = (const float*)d_in[19];
  const float* lat_w1   = (const float*)d_in[20];
  const float* lat_w2   = (const float*)d_in[21];
  const float* fin_norm = (const float*)d_in[22];
  const float* lm_head  = (const float*)d_in[23];
  float* logits = (float*)d_out;

  // ---- bump allocator over d_ws (~122 MB total) ----
  char* base = (char*)d_ws;
  size_t off = 0;
  auto alloc = [&](size_t elems) -> float* {
    off = (off + 255) & ~(size_t)255;
    float* p = (float*)(base + off);
    off += elems * sizeof(float);
    return p;
  };
  float* x      = alloc((size_t)Mfull * Dc);        // residual stream
  float* xn     = alloc((size_t)Mfull * Dc);        // norm output
  float* qkvbuf = alloc((size_t)Mfull * 3 * Dc);    // qkv / g_in / latent-hidden
  float* ctx    = alloc((size_t)Mfull * Dc);        // attn context / mem query
  float* proj   = alloc((size_t)Mfull * Dc);        // proj / latent out
  float* h1     = alloc((size_t)Mfull * DFFc);      // ffn h1 / retrieved
  float* h2     = alloc((size_t)Mfull * DFFc);      // ffn h2 / weighted values
  float* scores = alloc(Mfull);
  int*   tokidx = (int*)alloc(Bc * CAPc);
  float* xs     = alloc((size_t)Mcap * Dc);
  float* xsn    = alloc((size_t)Mcap * Dc);
  float* xsqkv  = alloc((size_t)Mcap * 3 * Dc);
  float* xsctx  = alloc((size_t)Mcap * Dc);
  float* xsproj = alloc((size_t)Mcap * Dc);
  float* xsh1   = alloc((size_t)Mcap * DFFc);
  float* xsh2   = alloc((size_t)Mcap * DFFc);
  float* sim    = alloc((size_t)Mfull * NSLOTSc);
  float* tkw    = alloc((size_t)Mfull * TOPKc);
  int*   tki    = (int*)alloc((size_t)Mfull * TOPKc);
  float* gh     = alloc((size_t)Mfull * (Dc / 2));
  float* gate   = alloc(Mfull);
  (void)ws_size; (void)n_in; (void)in_sizes; (void)out_size;

  auto gemm = [&](const float* A, const float* W, const float* bias, float* C,
                  int M, int N, int K, float alpha, int act) {
    dim3 g((N + GBN - 1) / GBN, (M + GBM - 1) / GBM);
    k_gemm_bf16<<<g, 256, 0, stream>>>(A, W, bias, C, M, N, K, alpha, act);
  };

  // ---- TDM / cluster instruction probe (architectural no-op) ----
  k_tdm_touch<<<1, 32, 0, stream>>>();

  // ---- embedding ----
  k_embed<<<Mfull, 256, 0, stream>>>(ids, embed, pos, x, Tc, Dc);

  // ---- transformer layers ----
  for (int i = 0; i < Lc; ++i) {
    const float* n1i  = n1 + (size_t)i * Dc;
    const float* qwi  = qkv_w + (size_t)i * 3 * Dc * Dc;
    const float* owi  = out_w + (size_t)i * Dc * Dc;
    const float* n2i  = n2 + (size_t)i * Dc;
    const float* w1i  = w1 + (size_t)i * DFFc * Dc;
    const float* w2i  = w2 + (size_t)i * DFFc * Dc;
    const float* w3i  = w3 + (size_t)i * Dc * DFFc;
    if (i & 1) {
      // Mixture-of-Depths: route top-CAP tokens
      k_router<<<Mfull, 256, 0, stream>>>(x, router + (size_t)i * Dc, scores, Dc);
      k_topk_tokens<<<Bc, 256, 0, stream>>>(scores, tokidx, Tc, CAPc);
      k_gather_rows<<<Mcap, 256, 0, stream>>>(x, tokidx, xs, Tc, CAPc, Dc);
      k_rmsnorm<<<Mcap, 256, 0, stream>>>(xs, n1i, xsn, Dc);
      gemm(xsn, qwi, nullptr, xsqkv, Mcap, 3 * Dc, Dc, 1.f, 0);
      k_attn<<<dim3(CAPc, Bc * Hc), 128, 0, stream>>>(xsqkv, xsctx, CAPc, Dc, Hc);
      gemm(xsctx, owi, nullptr, xsproj, Mcap, Dc, Dc, 1.f, 0);
      k_add<<<512, 256, 0, stream>>>(xs, xsproj, (size_t)Mcap * Dc);
      k_rmsnorm<<<Mcap, 256, 0, stream>>>(xs, n2i, xsn, Dc);
      gemm(xsn, w1i, nullptr, xsh1, Mcap, DFFc, Dc, 1.f, 1);   // silu
      gemm(xsn, w2i, nullptr, xsh2, Mcap, DFFc, Dc, 1.f, 0);
      k_mul<<<512, 256, 0, stream>>>(xsh1, xsh2, (size_t)Mcap * DFFc);
      gemm(xsh1, w3i, nullptr, xsproj, Mcap, Dc, DFFc, 1.f, 0);
      k_add<<<512, 256, 0, stream>>>(xs, xsproj, (size_t)Mcap * Dc);
      k_scatter_rows<<<Mcap, 256, 0, stream>>>(x, tokidx, xs, Tc, CAPc, Dc);
    } else {
      k_rmsnorm<<<Mfull, 256, 0, stream>>>(x, n1i, xn, Dc);
      gemm(xn, qwi, nullptr, qkvbuf, Mfull, 3 * Dc, Dc, 1.f, 0);
      k_attn<<<dim3(Tc, Bc * Hc), 128, 0, stream>>>(qkvbuf, ctx, Tc, Dc, Hc);
      gemm(ctx, owi, nullptr, proj, Mfull, Dc, Dc, 1.f, 0);
      k_add<<<2048, 256, 0, stream>>>(x, proj, (size_t)Mfull * Dc);
      k_rmsnorm<<<Mfull, 256, 0, stream>>>(x, n2i, xn, Dc);
      gemm(xn, w1i, nullptr, h1, Mfull, DFFc, Dc, 1.f, 1);     // silu
      gemm(xn, w2i, nullptr, h2, Mfull, DFFc, Dc, 1.f, 0);
      k_mul<<<2048, 256, 0, stream>>>(h1, h2, (size_t)Mfull * DFFc);
      gemm(h1, w3i, nullptr, proj, Mfull, Dc, DFFc, 1.f, 0);
      k_add<<<2048, 256, 0, stream>>>(x, proj, (size_t)Mfull * Dc);
    }
  }

  // ---- kNN memory retrieval ----
  gemm(x, mem_q, nullptr, ctx, Mfull, Dc, Dc, 1.f, 0);                       // q
  gemm(ctx, mem_keys, nullptr, sim, Mfull, NSLOTSc, Dc, 1.f / 32.f, 0);      // /sqrt(D)
  k_memtopk<<<(Mfull + 255) / 256, 256, 0, stream>>>(sim, tkw, tki, Mfull, NSLOTSc, TOPKc);
  k_retrieve<<<Mfull, 256, 0, stream>>>(tkw, tki, mem_vals, h2, Dc, TOPKc);  // weighted V
  gemm(h2, mem_out, nullptr, h1, Mfull, Dc, Dc, 1.f, 0);                     // retrieved
  k_concat<<<Mfull, 256, 0, stream>>>(x, h1, qkvbuf, Dc);                    // g_in
  gemm(qkvbuf, gate_w1, gate_b1, gh, Mfull, Dc / 2, 2 * Dc, 1.f, 2);         // gelu
  gemm(gh, gate_w2, gate_b2, gate, Mfull, 1, Dc / 2, 1.f, 3);                // sigmoid
  k_gatefuse<<<Mfull, 256, 0, stream>>>(x, gate, h1, Dc);

  // ---- latent reasoning iterations ----
  for (int it = 0; it < NTHc; ++it) {
    k_rmsnorm<<<Mfull, 256, 0, stream>>>(x, lat_norm, xn, Dc);
    gemm(xn, lat_w1, nullptr, qkvbuf, Mfull, 2 * Dc, Dc, 1.f, 2);            // gelu
    gemm(qkvbuf, lat_w2, nullptr, proj, Mfull, Dc, 2 * Dc, 1.f, 0);
    k_add<<<2048, 256, 0, stream>>>(x, proj, (size_t)Mfull * Dc);
  }

  // ---- final norm + LM head via TDM-staged WMMA GEMM (2048 x 50257 x 1024) ----
  k_rmsnorm<<<Mfull, 256, 0, stream>>>(x, fin_norm, xn, Dc);
  {
    dim3 g((Vc + GBN - 1) / GBN, (Mfull + GBM - 1) / GBM);
    k_gemm_bf16_tdm<<<g, 256, 0, stream>>>(xn, lm_head, logits, Mfull, Vc, Dc, 1.f);
  }
}
